// TransformerBlock_30709016166729
// MI455X (gfx1250) — compile-verified, matrix operand order checked by reading, not run on hardware
//
#include <hip/hip_runtime.h>

// ---------------------------------------------------------------------------
// Types / constants
// ---------------------------------------------------------------------------
typedef unsigned short u16;
typedef unsigned int   u32;
typedef __attribute__((ext_vector_type(16))) __bf16 v16bf;
typedef __attribute__((ext_vector_type(8)))  float  v8f;

#define D_MODEL   2048
#define D_FF      5632
#define NUM_HEADS 16
#define DK        128
#define SEQ       2048
#define BATCH     2
#define MTOK      (BATCH*SEQ)   // 4096 tokens

__device__ __forceinline__ u16 f2bf(float f) {
  u32 u = __float_as_uint(f);
  u32 r = u + 0x7fffu + ((u >> 16) & 1u);   // round-to-nearest-even
  return (u16)(r >> 16);
}
__device__ __forceinline__ float bf2f(u16 h) {
  return __uint_as_float(((u32)h) << 16);
}

union FragU { v16bf v; uint4 q[2]; u16 s[16]; };

__device__ __forceinline__ v8f zero8() {
  v8f z = {0.f,0.f,0.f,0.f,0.f,0.f,0.f,0.f};
  return z;
}

// Load a 16x32 bf16 WMMA A/B fragment from row-major (stride in halves) data.
// ISA layout: lane<16 -> row (base+lane),  K 0..7 (low 16B) & 16..23 (hi 16B)
//             lane>=16 -> row (base+lane-16), K 8..15 & 24..31
__device__ __forceinline__ v16bf load_frag(const u16* p, int rowBase, int stride, int lane) {
  int row  = rowBase + (lane & 15);
  int koff = (lane >> 4) << 3;               // 0 or 8
  FragU u;
  u.q[0] = *(const uint4*)(p + (size_t)row * stride + koff);
  u.q[1] = *(const uint4*)(p + (size_t)row * stride + koff + 16);
  return u.v;
}

// ---------------------------------------------------------------------------
// f32 -> bf16 weight/activation conversion (4 elems / thread)
// ---------------------------------------------------------------------------
__global__ __launch_bounds__(256) void cvt_bf16_kernel(const float* __restrict__ S,
                                                       u16* __restrict__ D) {
  int i = (blockIdx.x * 256 + threadIdx.x) * 4;
  float4 f = *(const float4*)(S + i);
  union { unsigned long long ll; u16 s[4]; } p;
  p.s[0] = f2bf(f.x); p.s[1] = f2bf(f.y); p.s[2] = f2bf(f.z); p.s[3] = f2bf(f.w);
  *(unsigned long long*)(D + i) = p.ll;
}

// ---------------------------------------------------------------------------
// RMSNorm: one block per token row (D_MODEL = 2048, 256 thr x 8 elems)
// ---------------------------------------------------------------------------
__global__ __launch_bounds__(256) void rmsnorm_kernel(const float* __restrict__ X,
                                                      const float* __restrict__ g,
                                                      u16* __restrict__ O) {
  int row = blockIdx.x, tid = threadIdx.x;
  const float* xr = X + (size_t)row * D_MODEL;
  float s = 0.f;
  #pragma unroll
  for (int e = 0; e < 8; ++e) { float v = xr[tid * 8 + e]; s += v * v; }
  __shared__ float red[256];
  red[tid] = s; __syncthreads();
  for (int o = 128; o > 0; o >>= 1) { if (tid < o) red[tid] += red[tid + o]; __syncthreads(); }
  float rinv = rsqrtf(red[0] * (1.0f / D_MODEL) + 1e-5f);
  #pragma unroll
  for (int e = 0; e < 8; ++e) {
    int c = tid * 8 + e;
    O[(size_t)row * D_MODEL + c] = f2bf(xr[c] * g[c] * rinv);
  }
}

// ---------------------------------------------------------------------------
// Generic NT GEMM:  C[M,N] = A[M,K] * B[N,K]^T   (A,B bf16, f32 accumulate)
// MODE 0: C -> bf16      MODE 1: C -> f32, += residual R
// 256 threads = 8 waves; block tile 128x128x32; wave tile 32x64 (8 WMMAs)
// ---------------------------------------------------------------------------
template<int MODE>
__global__ __launch_bounds__(256) void gemm_nt(const u16* __restrict__ A,
                                               const u16* __restrict__ B,
                                               u16* __restrict__ Cb,
                                               float* __restrict__ Cf,
                                               const float* __restrict__ R,
                                               int M, int N, int K) {
  const int tid = threadIdx.x, lane = tid & 31, wave = tid >> 5;
  const int wm = wave >> 1, wn = wave & 1;
  const int bm = blockIdx.y * 128, bn = blockIdx.x * 128;

  __shared__ alignas(16) u16 As[128 * 40];
  __shared__ alignas(16) u16 Bs[128 * 40];

  v8f acc[2][4];
  #pragma unroll
  for (int mt = 0; mt < 2; ++mt)
    #pragma unroll
    for (int nt = 0; nt < 4; ++nt) acc[mt][nt] = zero8();

  const int nk = K >> 5;
  for (int kt = 0; kt < nk; ++kt) {
    const int k0 = kt << 5;
    __syncthreads();
    #pragma unroll
    for (int i = tid; i < 512; i += 256) {
      int m = i >> 2, kc = (i & 3) << 3;
      *(uint4*)&As[m * 40 + kc] = *(const uint4*)&A[(size_t)(bm + m) * K + k0 + kc];
      *(uint4*)&Bs[m * 40 + kc] = *(const uint4*)&B[(size_t)(bn + m) * K + k0 + kc];
    }
    if (kt + 1 < nk) {                      // global_prefetch_b8 next K-slab
      int pr = tid & 127;
      __builtin_prefetch(&A[(size_t)(bm + pr) * K + k0 + 32], 0, 1);
      __builtin_prefetch(&B[(size_t)(bn + pr) * K + k0 + 32], 0, 1);
    }
    __syncthreads();

    v16bf af[2], bfr[4];
    #pragma unroll
    for (int mt = 0; mt < 2; ++mt) af[mt]  = load_frag(As, wm * 32 + mt * 16, 40, lane);
    #pragma unroll
    for (int nt = 0; nt < 4; ++nt) bfr[nt] = load_frag(Bs, wn * 64 + nt * 16, 40, lane);
    #pragma unroll
    for (int mt = 0; mt < 2; ++mt)
      #pragma unroll
      for (int nt = 0; nt < 4; ++nt)
        acc[mt][nt] = __builtin_amdgcn_wmma_f32_16x16x32_bf16(
            false, af[mt], false, bfr[nt], (short)0, acc[mt][nt], false, false);
  }

  const int rO = (lane >> 4) << 3;          // C layout: M = j + (lane>=16 ? 8 : 0)
  const int c0 = lane & 15;
  #pragma unroll
  for (int mt = 0; mt < 2; ++mt)
    #pragma unroll
    for (int nt = 0; nt < 4; ++nt)
      #pragma unroll
      for (int j = 0; j < 8; ++j) {
        int row = bm + wm * 32 + mt * 16 + rO + j;
        int col = bn + wn * 64 + nt * 16 + c0;
        size_t idx = (size_t)row * N + col;
        float v = acc[mt][nt][j];
        if (MODE == 0) Cb[idx] = f2bf(v);
        else           Cf[idx] = R[idx] + v;
      }
}

// ---------------------------------------------------------------------------
// RoPE on Q,K + transpose V into Vt[b,h,dk,s]. One thread per (b,s,h,pair).
// ---------------------------------------------------------------------------
__global__ __launch_bounds__(256) void rope_v_kernel(const u16* __restrict__ Ql,
                                                     const u16* __restrict__ Kl,
                                                     const u16* __restrict__ Vl,
                                                     u16* __restrict__ Qr,
                                                     u16* __restrict__ Kr,
                                                     u16* __restrict__ Vt) {
  int t = blockIdx.x * 256 + threadIdx.x;          // < 4194304
  int i = t & 63; int rem = t >> 6;
  int h = rem & 15; rem >>= 4;
  int s = rem & 2047; int b = rem >> 11;

  size_t lin = ((size_t)(b * SEQ + s)) * D_MODEL + h * DK + 2 * i;
  float q1 = bf2f(Ql[lin]), q2 = bf2f(Ql[lin + 1]);
  float k1 = bf2f(Kl[lin]), k2 = bf2f(Kl[lin + 1]);

  // inv_freq = theta^(-i/64), theta = 10000 -> ln(10000)=9.210340372
  float ang = (float)s * __expf(-(float)i * (9.21034037198f / 64.0f));
  float sn, cs; __sincosf(ang, &sn, &cs);

  size_t ro = ((size_t)((b * NUM_HEADS + h) * SEQ + s)) * DK + 2 * i;
  Qr[ro]     = f2bf(q1 * cs - q2 * sn);
  Qr[ro + 1] = f2bf(q1 * sn + q2 * cs);
  Kr[ro]     = f2bf(k1 * cs - k2 * sn);
  Kr[ro + 1] = f2bf(k1 * sn + k2 * cs);

  size_t vo = ((size_t)((b * NUM_HEADS + h) * DK + 2 * i)) * SEQ + s;
  Vt[vo]       = Vl[lin];
  Vt[vo + SEQ] = Vl[lin + 1];
}

// ---------------------------------------------------------------------------
// Flash attention: block = (b, h, 128 q rows); 8 waves, 16 q rows per wave.
// kv tiles of 32 shared via LDS; online softmax; ctx accumulated in v8f x 8.
// ---------------------------------------------------------------------------
__global__ __launch_bounds__(256) void attn_kernel(const u16* __restrict__ Qr,
                                                   const u16* __restrict__ Kr,
                                                   const u16* __restrict__ Vt,
                                                   u16* __restrict__ Ctx) {
  int bid = blockIdx.x;
  int qb = bid & 15, h = (bid >> 4) & 15, b = bid >> 8;
  int tid = threadIdx.x, lane = tid & 31, w = tid >> 5;
  int q0 = qb * 128 + w * 16;

  const size_t bh = (size_t)(b * NUM_HEADS + h);
  const u16* Qbase = Qr + bh * SEQ * DK;
  const u16* Kbase = Kr + bh * SEQ * DK;
  const u16* Vbase = Vt + bh * DK * SEQ;

  __shared__ alignas(16) u16 Ks[32 * 136];   // 32 kv rows x 128 dk (pad 136)
  __shared__ alignas(16) u16 Vs[128 * 40];   // 128 dk rows x 32 kv (pad 40)
  __shared__ float Ps[8][16][33];            // per-wave score/prob scratch
  __shared__ float Sc[8][16];                // per-wave row scale broadcast

  v16bf aq[4];
  #pragma unroll
  for (int c = 0; c < 4; ++c) aq[c] = load_frag(Qbase + c * 32, q0, DK, lane);

  v8f ctx[8];
  #pragma unroll
  for (int nt = 0; nt < 8; ++nt) ctx[nt] = zero8();

  float m_r = -1e30f, l_r = 0.f;
  const int r = lane & 15;
  const int rO = (lane >> 4) << 3;
  const int c0 = lane & 15;
  const int nk = qb * 4 + 4;                 // causal: kv <= block's max q
  const float scl = 0.08838834764831845f;    // 1/sqrt(128)

  for (int t = 0; t < nk; ++t) {
    const int k0 = t * 32;
    __syncthreads();
    #pragma unroll
    for (int i = tid; i < 512; i += 256) {   // K tile
      int rr = i >> 4, cc = (i & 15) << 3;
      *(uint4*)&Ks[rr * 136 + cc] = *(const uint4*)&Kbase[(size_t)(k0 + rr) * DK + cc];
    }
    #pragma unroll
    for (int i = tid; i < 512; i += 256) {   // V^T tile
      int rr = i >> 2, cc = (i & 3) << 3;
      *(uint4*)&Vs[rr * 40 + cc] = *(const uint4*)&Vbase[(size_t)rr * SEQ + k0 + cc];
    }
    __syncthreads();

    // scores: Q[16x128] * K^T -> 16x32 (2 tiles x 4 dk-chunk WMMAs)
    #pragma unroll
    for (int nt = 0; nt < 2; ++nt) {
      v8f s8 = zero8();
      #pragma unroll
      for (int c = 0; c < 4; ++c) {
        v16bf kf = load_frag(Ks + c * 32, nt * 16, 136, lane);
        s8 = __builtin_amdgcn_wmma_f32_16x16x32_bf16(
            false, aq[c], false, kf, (short)0, s8, false, false);
      }
      #pragma unroll
      for (int j = 0; j < 8; ++j) {
        int rrow = rO + j, ccol = nt * 16 + c0;
        float v = s8[j] * scl;
        if (k0 + ccol > q0 + rrow) v = -1e30f;   // causal mask
        Ps[w][rrow][ccol] = v;
      }
    }
    __syncthreads();

    // online softmax (lane pairs L and L+16 both process row L&15)
    float nm = m_r;
    #pragma unroll
    for (int c = 0; c < 32; ++c) nm = fmaxf(nm, Ps[w][r][c]);
    float alpha = __expf(m_r - nm);
    float sum = 0.f;
    #pragma unroll
    for (int c = 0; c < 32; ++c) {
      float p = __expf(Ps[w][r][c] - nm);
      sum += p;
      if (lane < 16) Ps[w][r][c] = p;
    }
    l_r = l_r * alpha + sum; m_r = nm;
    if (lane < 16) Sc[w][r] = alpha;
    __syncthreads();

    // rescale ctx accumulators
    #pragma unroll
    for (int j = 0; j < 8; ++j) {
      float al = Sc[w][rO + j];
      #pragma unroll
      for (int nt = 0; nt < 8; ++nt) ctx[nt][j] *= al;
    }

    // build P fragment (A-layout) from probs
    FragU pu;
    #pragma unroll
    for (int e = 0; e < 8; ++e) {
      pu.s[e]     = f2bf(Ps[w][r][rO + e]);
      pu.s[e + 8] = f2bf(Ps[w][r][rO + 16 + e]);
    }

    // ctx += P[16x32] * V[32x128]  (8 WMMAs, B-frags from Vt rows)
    #pragma unroll
    for (int nt = 0; nt < 8; ++nt) {
      v16bf vf = load_frag(Vs, nt * 16, 40, lane);
      ctx[nt] = __builtin_amdgcn_wmma_f32_16x16x32_bf16(
          false, pu.v, false, vf, (short)0, ctx[nt], false, false);
    }
  }

  __syncthreads();
  if (lane < 16) Sc[w][r] = 1.0f / l_r;
  __syncthreads();

  #pragma unroll
  for (int j = 0; j < 8; ++j) {
    float iv = Sc[w][rO + j];
    int srow = q0 + rO + j;
    #pragma unroll
    for (int nt = 0; nt < 8; ++nt) {
      int d = nt * 16 + c0;
      Ctx[(size_t)(b * SEQ + srow) * D_MODEL + h * DK + d] = f2bf(ctx[nt][j] * iv);
    }
  }
}

// ---------------------------------------------------------------------------
// SwiGLU elementwise: h = silu(a) * gate
// ---------------------------------------------------------------------------
__global__ __launch_bounds__(256) void silu_gate_kernel(const u16* __restrict__ A,
                                                        const u16* __restrict__ G,
                                                        u16* __restrict__ H) {
  int i = blockIdx.x * 256 + threadIdx.x;
  float x = bf2f(A[i]);
  float s = x / (1.0f + __expf(-x));
  H[i] = f2bf(s * bf2f(G[i]));
}

// ---------------------------------------------------------------------------
// Workspace layout (bytes)
// ---------------------------------------------------------------------------
static constexpr size_t SZ_W   = (size_t)D_MODEL * D_MODEL * 2;   //  8 MB
static constexpr size_t SZ_WF  = (size_t)D_FF * D_MODEL * 2;      // 22 MB
static constexpr size_t SZ_ACT = (size_t)MTOK * D_MODEL * 2;      // 16 MB
static constexpr size_t SZ_FF  = (size_t)MTOK * D_FF * 2;         // 44 MB

static constexpr size_t OFF_WQ  = 0;
static constexpr size_t OFF_WK  = OFF_WQ + SZ_W;
static constexpr size_t OFF_WV  = OFF_WK + SZ_W;
static constexpr size_t OFF_WO  = OFF_WV + SZ_W;
static constexpr size_t OFF_W1  = OFF_WO + SZ_W;
static constexpr size_t OFF_W2  = OFF_W1 + SZ_WF;
static constexpr size_t OFF_W3  = OFF_W2 + SZ_WF;
static constexpr size_t OFF_XN  = OFF_W3 + SZ_WF;
static constexpr size_t OFF_QL  = OFF_XN + SZ_ACT;   // reused later as ABUF
static constexpr size_t OFF_KL  = OFF_QL + SZ_ACT;
static constexpr size_t OFF_VL  = OFF_KL + SZ_ACT;
static constexpr size_t OFF_QR  = OFF_VL + SZ_ACT;   // reused later as GBUF
static constexpr size_t OFF_KR  = OFF_QR + SZ_ACT;
static constexpr size_t OFF_VT  = OFF_KR + SZ_ACT;
static constexpr size_t OFF_CTX = OFF_VT + SZ_ACT;
static constexpr size_t OFF_X1  = OFF_CTX + SZ_ACT;  // f32 residual, 32 MB
static constexpr size_t OFF_XN2 = OFF_X1 + (size_t)MTOK * D_MODEL * 4;
static constexpr size_t OFF_H   = OFF_XN2 + SZ_ACT;
static constexpr size_t OFF_AB  = OFF_QL;            // a buffer   (46 MB fits QL+KL+VL)
static constexpr size_t OFF_GB  = OFF_QR;            // gate buffer (fits QR+KR+VT)

extern "C" void kernel_launch(void* const* d_in, const int* in_sizes, int n_in,
                              void* d_out, int out_size, void* d_ws, size_t ws_size,
                              hipStream_t stream) {
  (void)in_sizes; (void)n_in; (void)out_size; (void)ws_size;

  const float* x  = (const float*)d_in[0];
  const float* Wq = (const float*)d_in[1];
  const float* Wk = (const float*)d_in[2];
  const float* Wv = (const float*)d_in[3];
  const float* Wo = (const float*)d_in[4];
  const float* W1 = (const float*)d_in[5];
  const float* W2 = (const float*)d_in[6];
  const float* W3 = (const float*)d_in[7];
  const float* g1 = (const float*)d_in[8];
  const float* g2 = (const float*)d_in[9];
  float* out = (float*)d_out;

  char* ws = (char*)d_ws;
  u16*   wq  = (u16*)(ws + OFF_WQ);
  u16*   wk  = (u16*)(ws + OFF_WK);
  u16*   wv  = (u16*)(ws + OFF_WV);
  u16*   wo  = (u16*)(ws + OFF_WO);
  u16*   w1  = (u16*)(ws + OFF_W1);
  u16*   w2  = (u16*)(ws + OFF_W2);
  u16*   w3  = (u16*)(ws + OFF_W3);
  u16*   xn  = (u16*)(ws + OFF_XN);
  u16*   ql  = (u16*)(ws + OFF_QL);
  u16*   kl  = (u16*)(ws + OFF_KL);
  u16*   vl  = (u16*)(ws + OFF_VL);
  u16*   qr  = (u16*)(ws + OFF_QR);
  u16*   kr  = (u16*)(ws + OFF_KR);
  u16*   vt  = (u16*)(ws + OFF_VT);
  u16*   ctx = (u16*)(ws + OFF_CTX);
  float* x1  = (float*)(ws + OFF_X1);
  u16*   xn2 = (u16*)(ws + OFF_XN2);
  u16*   ab  = (u16*)(ws + OFF_AB);
  u16*   gb  = (u16*)(ws + OFF_GB);
  u16*   hb  = (u16*)(ws + OFF_H);

  // 1) weights -> bf16
  cvt_bf16_kernel<<<(D_MODEL*D_MODEL)/1024, 256, 0, stream>>>(Wq, wq);
  cvt_bf16_kernel<<<(D_MODEL*D_MODEL)/1024, 256, 0, stream>>>(Wk, wk);
  cvt_bf16_kernel<<<(D_MODEL*D_MODEL)/1024, 256, 0, stream>>>(Wv, wv);
  cvt_bf16_kernel<<<(D_MODEL*D_MODEL)/1024, 256, 0, stream>>>(Wo, wo);
  cvt_bf16_kernel<<<(D_FF*D_MODEL)/1024,    256, 0, stream>>>(W1, w1);
  cvt_bf16_kernel<<<(D_FF*D_MODEL)/1024,    256, 0, stream>>>(W2, w2);
  cvt_bf16_kernel<<<(D_FF*D_MODEL)/1024,    256, 0, stream>>>(W3, w3);

  // 2) xn = RMSNorm(x, g1)
  rmsnorm_kernel<<<MTOK, 256, 0, stream>>>(x, g1, xn);

  // 3) QKV projections (WMMA GEMMs)
  dim3 gQKV(D_MODEL/128, MTOK/128);
  gemm_nt<0><<<gQKV, 256, 0, stream>>>(xn, wq, ql, nullptr, nullptr, MTOK, D_MODEL, D_MODEL);
  gemm_nt<0><<<gQKV, 256, 0, stream>>>(xn, wk, kl, nullptr, nullptr, MTOK, D_MODEL, D_MODEL);
  gemm_nt<0><<<gQKV, 256, 0, stream>>>(xn, wv, vl, nullptr, nullptr, MTOK, D_MODEL, D_MODEL);

  // 4) RoPE(Q,K) + V transpose
  rope_v_kernel<<<(BATCH*SEQ*NUM_HEADS*(DK/2))/256, 256, 0, stream>>>(ql, kl, vl, qr, kr, vt);

  // 5) flash attention -> ctx[b,s,h,dk]
  attn_kernel<<<BATCH*NUM_HEADS*(SEQ/128), 256, 0, stream>>>(qr, kr, vt, ctx);

  // 6) x1 = x + ctx * Wo^T
  gemm_nt<1><<<gQKV, 256, 0, stream>>>(ctx, wo, nullptr, x1, x, MTOK, D_MODEL, D_MODEL);

  // 7) xn2 = RMSNorm(x1, g2)
  rmsnorm_kernel<<<MTOK, 256, 0, stream>>>(x1, g2, xn2);

  // 8) a = xn2*W1^T, gate = xn2*W3^T
  dim3 gFF(D_FF/128, MTOK/128);
  gemm_nt<0><<<gFF, 256, 0, stream>>>(xn2, w1, ab, nullptr, nullptr, MTOK, D_FF, D_MODEL);
  gemm_nt<0><<<gFF, 256, 0, stream>>>(xn2, w3, gb, nullptr, nullptr, MTOK, D_FF, D_MODEL);

  // 9) h = silu(a) * gate
  silu_gate_kernel<<<(MTOK*D_FF)/256, 256, 0, stream>>>(ab, gb, hb);

  // 10) out = x1 + h * W2^T
  gemm_nt<1><<<gQKV, 256, 0, stream>>>(hb, w2, nullptr, out, x1, MTOK, D_MODEL, D_FF);
}
